// VectorQuantizer_22522808500718
// MI455X (gfx1250) — matrix operand order repaired and checked by reading, not compiled, not run on hardware
//
#include <hip/hip_runtime.h>

#if __has_builtin(__builtin_amdgcn_global_load_async_to_lds_b128)
#define VQ_ASYNC 1
#else
#define VQ_ASYNC 0
#endif

typedef __attribute__((ext_vector_type(16))) __bf16 v16bf;
typedef __attribute__((ext_vector_type(8)))  __bf16 v8bf;
typedef __attribute__((ext_vector_type(8)))  float  v8f;
typedef int v4i __attribute__((vector_size(4 * sizeof(int))));
typedef unsigned short u16;
typedef unsigned int   u32;

#define GPTR(p) ((__attribute__((address_space(1))) v4i*)(p))
#define LPTR(p) ((__attribute__((address_space(3))) v4i*)(p))

#define BATCH        32768
#define NUM_CODES    8192
#define NUM_CHANNELS 512

#define BM 128
#define BN 256
#define BK 32
#define KSTEPS (NUM_CHANNELS / BK)   // 16

__device__ __forceinline__ void vq_wait_async0() {
#if __has_builtin(__builtin_amdgcn_s_wait_asynccnt)
    __builtin_amdgcn_s_wait_asynccnt(0);
#else
    asm volatile("s_wait_asynccnt 0" ::: "memory");
#endif
}

// ---------------------------------------------------------------------------
// fp32 -> bf16 (RNE) helper
// ---------------------------------------------------------------------------
__device__ __forceinline__ u16 f2bf(float f) {
    u32 x = __float_as_uint(f);
    u32 lsb = (x >> 16) & 1u;
    x += 0x7fffu + lsb;          // round to nearest even
    return (u16)(x >> 16);
}

// ---------------------------------------------------------------------------
// Kernel 1/2: convert one 512-wide row to bf16 and compute its squared norm.
// One wave (32 lanes) per row, 8 rows per 256-thread block.
// ---------------------------------------------------------------------------
__global__ __launch_bounds__(256) void vq_cvt_rows(
    const float* __restrict__ src, u16* __restrict__ dst,
    float* __restrict__ sq)
{
    const int row  = blockIdx.x * 8 + (threadIdx.x >> 5);
    const int lane = threadIdx.x & 31;

    const float4* s = (const float4*)(src + (size_t)row * NUM_CHANNELS);
    u16* drow = dst + (size_t)row * NUM_CHANNELS;

    float acc = 0.f;
#pragma unroll
    for (int i = 0; i < 4; ++i) {
        const int idx4 = lane + 32 * i;           // coalesced float4 across wave
        float4 v = s[idx4];
        acc = fmaf(v.x, v.x, acc);
        acc = fmaf(v.y, v.y, acc);
        acc = fmaf(v.z, v.z, acc);
        acc = fmaf(v.w, v.w, acc);
        ushort4 o;
        o.x = f2bf(v.x); o.y = f2bf(v.y); o.z = f2bf(v.z); o.w = f2bf(v.w);
        *(ushort4*)(drow + idx4 * 4) = o;         // coalesced 8B stores
    }
#pragma unroll
    for (int off = 16; off > 0; off >>= 1)
        acc += __shfl_xor(acc, off, 32);
    if (lane == 0) sq[row] = acc;
}

// ---------------------------------------------------------------------------
// Kernel 3: tiled bf16 WMMA GEMM with fused VQ epilogue.
//   out[b][c] = 2 * sum_d K[b][d]*E[c][d]  -  k_sq[b]  -  e_sq[c]
// Block: 256 threads (8 waves), tile 128(M) x 256(N), K step 32,
// double-buffered LDS filled by ASYNC global->LDS DMA (ASYNCcnt),
// fully unrolled K-loop so stage offsets are compile-time constants.
// Each wave owns a 64x64 sub-tile = 4x4 WMMA frags.
// ---------------------------------------------------------------------------
__global__ __launch_bounds__(256)
__attribute__((amdgpu_waves_per_eu(2, 4)))
void vq_logits_wmma(
    const u16* __restrict__ keysbf,   // [BATCH][512]  bf16 bits
    const u16* __restrict__ embbf,    // [NUM_CODES][512] bf16 bits
    const float* __restrict__ k_sq,   // [BATCH]
    const float* __restrict__ e_sq,   // [NUM_CODES]
    float* __restrict__ out)          // [BATCH][NUM_CODES]
{
    __shared__ __align__(32) u16 As[2][BM * BK];   //  8 KB per stage
    __shared__ __align__(32) u16 Bs[2][BN * BK];   // 16 KB per stage

    const int tid   = threadIdx.x;
    const int wave  = tid >> 5;
    const int lane  = tid & 31;
    const int wm    = wave & 1;        // wave grid 2 (M) x 4 (N)
    const int wn    = wave >> 1;
    const int mo    = wm * 64;         // wave sub-tile origin within tile
    const int no    = wn * 64;
    const int l     = lane & 15;
    const int h     = lane >> 4;

    const int tileM = blockIdx.y * BM;
    const int tileN = blockIdx.x * BN;

    // fp32 accumulators: 4 (M frags) x 4 (N frags) = 128 VGPRs
    v8f acc[4][4];
#pragma unroll
    for (int mf = 0; mf < 4; ++mf)
#pragma unroll
        for (int nf = 0; nf < 4; ++nf)
#pragma unroll
            for (int i = 0; i < 8; ++i) acc[mf][nf][i] = 0.f;

    // A tile: 512 x 16B chunks (2/thread); B tile: 1024 x 16B chunks (4/thread)
    const int ldrow = tid >> 2;        // 0..63
    const int ldoff = tid & 3;         // 16B chunk within a 64B row

#if VQ_ASYNC
    // ---- CDNA5 async DMA path: global -> LDS, tracked by ASYNCcnt ----
    auto fill_stage = [&](int s, int k0) {
#pragma unroll
        for (int c = 0; c < 2; ++c) {
            const int row = ldrow + 64 * c;
            const u16* g = keysbf + (size_t)(tileM + row) * NUM_CHANNELS
                         + k0 + ldoff * 8;
            u16* lp = &As[s][row * BK + ldoff * 8];
            __builtin_amdgcn_global_load_async_to_lds_b128(
                GPTR(g), LPTR(lp), 0, 0);
        }
#pragma unroll
        for (int c = 0; c < 4; ++c) {
            const int row = ldrow + 64 * c;
            const u16* g = embbf + (size_t)(tileN + row) * NUM_CHANNELS
                         + k0 + ldoff * 8;
            u16* lp = &Bs[s][row * BK + ldoff * 8];
            __builtin_amdgcn_global_load_async_to_lds_b128(
                GPTR(g), LPTR(lp), 0, 0);
        }
    };

    fill_stage(0, 0);
    vq_wait_async0();
    __syncthreads();
#else
    // ---- fallback: register-staged copy ----
    uint4 ra[2], rb[4];
    auto load_regs = [&](int k0) {
#pragma unroll
        for (int c = 0; c < 2; ++c) {
            const int row = ldrow + 64 * c;
            const u16* pa = keysbf + (size_t)(tileM + row) * NUM_CHANNELS + k0;
            ra[c] = *((const uint4*)pa + ldoff);
        }
#pragma unroll
        for (int c = 0; c < 4; ++c) {
            const int row = ldrow + 64 * c;
            const u16* pb = embbf + (size_t)(tileN + row) * NUM_CHANNELS + k0;
            rb[c] = *((const uint4*)pb + ldoff);
        }
    };
    auto store_lds = [&](int s) {
#pragma unroll
        for (int c = 0; c < 2; ++c)
            *((uint4*)&As[s][(ldrow + 64 * c) * BK] + ldoff) = ra[c];
#pragma unroll
        for (int c = 0; c < 4; ++c)
            *((uint4*)&Bs[s][(ldrow + 64 * c) * BK] + ldoff) = rb[c];
    };

    load_regs(0);
    store_lds(0);
    __syncthreads();
#endif

#pragma unroll   // full unroll: stage offsets become compile-time constants
    for (int kt = 0; kt < KSTEPS; ++kt) {
        const int s = kt & 1;
#if VQ_ASYNC
        if (kt + 1 < KSTEPS) fill_stage(s ^ 1, (kt + 1) * BK);  // DMA overlaps
#else
        if (kt + 1 < KSTEPS) load_regs((kt + 1) * BK);
#endif

        // B fragments: contiguous 32B per lane (column N = l, K = 16h..16h+15)
        v16bf b[4];
#pragma unroll
        for (int nf = 0; nf < 4; ++nf)
            b[nf] = *(const v16bf*)&Bs[s][(no + 16 * nf + l) * BK + h * 16];

#pragma unroll
        for (int mf = 0; mf < 4; ++mf) {
            // A fragment: two 16B chunks (K 16h..16h+7 and 16h+8..16h+15)
            const v8bf* p =
                (const v8bf*)&As[s][(mo + 16 * mf + l) * BK + h * 16];
            v16bf a = __builtin_shufflevector(p[0], p[1],
                        0, 1, 2, 3, 4, 5, 6, 7, 8, 9, 10, 11, 12, 13, 14, 15);
#pragma unroll
            for (int nf = 0; nf < 4; ++nf)
                acc[mf][nf] = __builtin_amdgcn_wmma_f32_16x16x32_bf16(
                    false, a, false, b[nf], (short)0, acc[mf][nf],
                    false, false);
        }

#if VQ_ASYNC
        if (kt + 1 < KSTEPS) vq_wait_async0();   // my transfers into s^1 done
#else
        if (kt + 1 < KSTEPS) store_lds(s ^ 1);
#endif
        __syncthreads();                         // everyone's transfers visible
    }

    // --- fused epilogue: 2*acc - k_sq - e_sq ---
    // C/D layout: VGPR r holds M = r + 8h, N = lane%16.
#pragma unroll
    for (int mf = 0; mf < 4; ++mf) {
        const int rbase = tileM + mo + 16 * mf + 8 * h;
        float ks[8];
#pragma unroll
        for (int r = 0; r < 8; ++r) ks[r] = k_sq[rbase + r];
#pragma unroll
        for (int nf = 0; nf < 4; ++nf) {
            const int col = tileN + no + 16 * nf + l;
            const float es = e_sq[col];
#pragma unroll
            for (int r = 0; r < 8; ++r)
                out[(size_t)(rbase + r) * NUM_CODES + col] =
                    2.f * acc[mf][nf][r] - ks[r] - es;
        }
    }
}

// ---------------------------------------------------------------------------
extern "C" void kernel_launch(void* const* d_in, const int* in_sizes, int n_in,
                              void* d_out, int out_size, void* d_ws, size_t ws_size,
                              hipStream_t stream) {
    (void)in_sizes; (void)n_in; (void)out_size; (void)ws_size;

    const float* keys = (const float*)d_in[0];   // [32768, 512]
    const float* emb  = (const float*)d_in[1];   // [8192, 512]
    float* out = (float*)d_out;                  // [32768, 8192]

    // workspace layout
    char* ws = (char*)d_ws;
    u16*   keysbf = (u16*)ws;                                        // 33.55 MB
    u16*   embbf  = (u16*)(ws + (size_t)BATCH * NUM_CHANNELS * 2);   //  8.39 MB
    float* k_sq   = (float*)(ws + (size_t)(BATCH + NUM_CODES) * NUM_CHANNELS * 2);
    float* e_sq   = k_sq + BATCH;

    vq_cvt_rows<<<BATCH / 8,     256, 0, stream>>>(keys, keysbf, k_sq);
    vq_cvt_rows<<<NUM_CODES / 8, 256, 0, stream>>>(emb,  embbf,  e_sq);

    dim3 grid(NUM_CODES / BN, BATCH / BM);   // (32, 256)
    vq_logits_wmma<<<grid, 256, 0, stream>>>(keysbf, embbf, k_sq, e_sq, out);
}
